// LAS_42167988912851
// MI455X (gfx1250) — matrix operand order repaired and checked by reading, not compiled, not run on hardware
//
#include <hip/hip_runtime.h>
#include <hip/hip_bf16.h>

// ---------------------------------------------------------------------------
// Types
// ---------------------------------------------------------------------------
typedef __bf16 bf16;
typedef __attribute__((ext_vector_type(16))) __bf16 v16bf;
typedef __attribute__((ext_vector_type(8)))  __bf16 v8bf;
typedef __attribute__((ext_vector_type(8)))  float  v8f;

__device__ __forceinline__ float bf2f(bf16 x) {
  union { unsigned short u; bf16 b; } c; c.b = x;
  unsigned int t = ((unsigned int)c.u) << 16;
  return __uint_as_float(t);
}
__device__ __forceinline__ bf16 f2bf(float f) {
  unsigned int u = __float_as_uint(f);
  unsigned int r = u + 0x7FFFu + ((u >> 16) & 1u);
  union { unsigned short u; bf16 b; } c; c.u = (unsigned short)(r >> 16);
  return c.b;
}
__device__ __forceinline__ float sigf(float x) { return 1.f / (1.f + __expf(-x)); }

// ---------------------------------------------------------------------------
// WMMA accumulate: acc(32x16) += A(32xK row-major) @ W(rows n..n+15 of NxK)^T
// A fragment: lane m=l%16, k = 16*(l/16) + e              -> two adjacent 8-half runs
// B fragment: lane n=l%16, k = 16*(l/16) + e              -> one 16-half run
// ---------------------------------------------------------------------------
__device__ __forceinline__ void accum_pair(
    v8f& acc0, v8f& acc1,
    const bf16* __restrict__ A, int lda,
    const bf16* __restrict__ W, long wrow, int ldw, int K,
    int lm, int hi)
{
  const bf16* Ar0 = A + (long)lm * lda;
  const bf16* Ar1 = A + (long)(16 + lm) * lda;
  const bf16* Wr  = W + (wrow + lm) * (long)ldw + hi * 16;
  int aoff = hi * 16;
  for (int kk = 0; kk < K; kk += 32) {
    v8bf a0lo = *(const v8bf*)(Ar0 + kk + aoff);
    v8bf a0hi = *(const v8bf*)(Ar0 + kk + 8 + aoff);
    v8bf a1lo = *(const v8bf*)(Ar1 + kk + aoff);
    v8bf a1hi = *(const v8bf*)(Ar1 + kk + 8 + aoff);
    v16bf b   = *(const v16bf*)(Wr + kk);
    v16bf a0  = __builtin_shufflevector(a0lo, a0hi, 0,1,2,3,4,5,6,7,8,9,10,11,12,13,14,15);
    v16bf a1  = __builtin_shufflevector(a1lo, a1hi, 0,1,2,3,4,5,6,7,8,9,10,11,12,13,14,15);
    acc0 = __builtin_amdgcn_wmma_f32_16x16x32_bf16(false, a0, false, b, (short)0, acc0, false, false);
    acc1 = __builtin_amdgcn_wmma_f32_16x16x32_bf16(false, a1, false, b, (short)0, acc1, false, false);
  }
}

// ---------------------------------------------------------------------------
// Shared LSTM gate finish: gates -> LDS -> i,f,g,o nonlinearity -> c (f32), h (bf16)
// wave = gate index (i,f,g,o); gcol = wave*H + n0
// ---------------------------------------------------------------------------
__device__ __forceinline__ void gates_finish(
    float g[4][32][16],
    const v8f& acc0, const v8f& acc1,
    const bf16* __restrict__ Gpre, int ldg,
    const float* __restrict__ b0, const float* __restrict__ b1,
    int wave, int lm, int hi, int gcol,
    float* __restrict__ c, bf16* __restrict__ h, int H, int n0)
{
  float bs = 0.f;
  if (b0) bs += b0[gcol + lm];
  if (b1) bs += b1[gcol + lm];
#pragma unroll
  for (int r = 0; r < 8; ++r) {
    int m = r + 8 * hi;
    float v0 = acc0[r] + bs;
    float v1 = acc1[r] + bs;
    if (Gpre) {
      v0 += bf2f(Gpre[(long)m * ldg + gcol + lm]);
      v1 += bf2f(Gpre[(long)(m + 16) * ldg + gcol + lm]);
    }
    g[wave][m][lm]      = v0;
    g[wave][m + 16][lm] = v1;
  }
  __syncthreads();
  for (int idx = (int)threadIdx.x; idx < 512; idx += 128) {
    int b = idx >> 4, nn = idx & 15;
    float iv = g[0][b][nn], fv = g[1][b][nn], gv = g[2][b][nn], ov = g[3][b][nn];
    long ci = (long)b * H + n0 + nn;
    float cn = sigf(fv) * c[ci] + sigf(iv) * tanhf(gv);
    c[ci] = cn;
    h[ci] = f2bf(sigf(ov) * tanhf(cn));
  }
}

// ---------------------------------------------------------------------------
// Generic GEMM: C(MxN) = A(MxK,row) @ W(NxK,row)^T + bias0 + bias1
// block = 4 waves; wave -> 16 N-cols; block -> 32 M-rows x 64 N-cols
// M must be a multiple of 32; K multiple of 32; W allocated out to Npad rows.
// ---------------------------------------------------------------------------
template <bool BF16OUT>
__global__ void __launch_bounds__(128) wmma_gemm(
    const bf16* __restrict__ A, int lda,
    const bf16* __restrict__ W, int ldw,
    const float* __restrict__ bias0, const float* __restrict__ bias1,
    void* __restrict__ Cout, long ldc,
    int K, int N, int Npad)
{
  int wave = threadIdx.x >> 5;
  int lane = threadIdx.x & 31;
  int lm = lane & 15, hi = lane >> 4;
  int n0 = blockIdx.x * 64 + wave * 16;
  if (n0 >= Npad) return;                       // wave-uniform
  long m0 = (long)blockIdx.y * 32;
  v8f acc0 = {0.f,0.f,0.f,0.f,0.f,0.f,0.f,0.f};
  v8f acc1 = {0.f,0.f,0.f,0.f,0.f,0.f,0.f,0.f};
  accum_pair(acc0, acc1, A + m0 * lda, lda, W, n0, ldw, K, lm, hi);
  int n = n0 + lm;
  if (n < N) {
    float bs = 0.f;
    if (bias0) bs += bias0[n];
    if (bias1) bs += bias1[n];
#pragma unroll
    for (int r = 0; r < 8; ++r) {
      long mA = m0 + r + 8 * hi;
      long mB = m0 + 16 + r + 8 * hi;
      float v0 = acc0[r] + bs, v1 = acc1[r] + bs;
      if (BF16OUT) {
        ((bf16*)Cout)[mA * ldc + n] = f2bf(v0);
        ((bf16*)Cout)[mB * ldc + n] = f2bf(v1);
      } else {
        ((float*)Cout)[mA * ldc + n] = v0;
        ((float*)Cout)[mB * ldc + n] = v1;
      }
    }
  }
}

// ---------------------------------------------------------------------------
// Encoder pBLSTM step: both directions in one launch (grid.y = dir)
// gates = Gx[t] + h_prev @ Whh^T  (biases folded into Gx)
// ---------------------------------------------------------------------------
struct EncStepArgs {
  const bf16 *GxF, *GxB, *WhhF, *WhhB, *hzero;
  bf16 *hF, *hB;
  float *cF, *cB;
  int H, T, s;
};

__global__ void __launch_bounds__(128) enc_step(EncStepArgs a)
{
  __shared__ float g[4][32][16];
  int dir  = blockIdx.y;
  int wave = threadIdx.x >> 5, lane = threadIdx.x & 31;
  int lm = lane & 15, hi = lane >> 4;
  int n0 = blockIdx.x * 16;
  int H = a.H;
  int t = dir ? (a.T - 1 - a.s) : a.s;
  const bf16* Gx = dir ? a.GxB : a.GxF;
  const bf16* W  = dir ? a.WhhB : a.WhhF;
  bf16* hseq = dir ? a.hB : a.hF;
  float* c   = dir ? a.cB : a.cF;
  int tprev = dir ? t + 1 : t - 1;
  const bf16* hprev = (a.s == 0) ? a.hzero : hseq + (long)tprev * 32 * H;
  bf16* hout = hseq + (long)t * 32 * H;
  const bf16* Gpre = Gx + (long)t * 32 * (4 * H);
  long wrow = (long)wave * H + n0;
  v8f acc0 = {0.f,0.f,0.f,0.f,0.f,0.f,0.f,0.f};
  v8f acc1 = {0.f,0.f,0.f,0.f,0.f,0.f,0.f,0.f};
  accum_pair(acc0, acc1, hprev, H, W, wrow, H, H, lm, hi);
  gates_finish(g, acc0, acc1, Gpre, 4 * H, nullptr, nullptr,
               wave, lm, hi, (int)wrow, c, hout, H, n0);
}

// ---------------------------------------------------------------------------
// Decoder LSTM cell step: up to 3 (A,W) pairs + optional precomputed Gpre
// ---------------------------------------------------------------------------
struct DecStepArgs {
  const bf16 *A0, *W0; int lda0, ldw0, K0;
  const bf16 *A1, *W1; int lda1, ldw1, K1;
  const bf16 *A2, *W2; int lda2, ldw2, K2;
  const bf16 *Gpre; int ldg;
  const float *b0, *b1;
  float *c; bf16 *h; int H;
};

__global__ void __launch_bounds__(128) dec_step(DecStepArgs a)
{
  __shared__ float g[4][32][16];
  int wave = threadIdx.x >> 5, lane = threadIdx.x & 31;
  int lm = lane & 15, hi = lane >> 4;
  int n0 = blockIdx.x * 16;
  long wrow = (long)wave * a.H + n0;
  v8f acc0 = {0.f,0.f,0.f,0.f,0.f,0.f,0.f,0.f};
  v8f acc1 = {0.f,0.f,0.f,0.f,0.f,0.f,0.f,0.f};
  if (a.A0) accum_pair(acc0, acc1, a.A0, a.lda0, a.W0, wrow, a.ldw0, a.K0, lm, hi);
  if (a.A1) accum_pair(acc0, acc1, a.A1, a.lda1, a.W1, wrow, a.ldw1, a.K1, lm, hi);
  if (a.A2) accum_pair(acc0, acc1, a.A2, a.lda2, a.W2, wrow, a.ldw2, a.K2, lm, hi);
  gates_finish(g, acc0, acc1, a.Gpre, a.ldg, a.b0, a.b1,
               wave, lm, hi, (int)wrow, a.c, a.h, a.H, n0);
}

// ---------------------------------------------------------------------------
// Fused attention per batch row: scores = q . keys[t], softmax, ctx = attn @ values
// keys/values stored as (Tk*B, HD) with row index t*32 + b. Tk = 128, HD = 256.
// ---------------------------------------------------------------------------
__global__ void __launch_bounds__(128) attention(
    const float* __restrict__ q, const float* __restrict__ keys,
    const float* __restrict__ values, bf16* __restrict__ ctx)
{
  __shared__ float sq[256];
  __shared__ float sc[128];
  __shared__ float red[128];
  int b = blockIdx.x, tid = threadIdx.x;
  for (int i = tid; i < 256; i += 128) sq[i] = q[b * 256 + i];
  __syncthreads();
  const float* kr = keys + ((long)tid * 32 + b) * 256;
  float s = 0.f;
  for (int d = 0; d < 256; ++d) s += sq[d] * kr[d];
  sc[tid] = s; red[tid] = s;
  __syncthreads();
  for (int st = 64; st > 0; st >>= 1) {
    if (tid < st) red[tid] = fmaxf(red[tid], red[tid + st]);
    __syncthreads();
  }
  float mx = red[0];
  __syncthreads();
  float e = __expf(sc[tid] - mx);
  sc[tid] = e; red[tid] = e;
  __syncthreads();
  for (int st = 64; st > 0; st >>= 1) {
    if (tid < st) red[tid] += red[tid + st];
    __syncthreads();
  }
  float inv = 1.f / red[0];
  __syncthreads();
  sc[tid] *= inv;
  __syncthreads();
  for (int d = tid; d < 256; d += 128) {
    float a = 0.f;
    for (int t = 0; t < 128; ++t) a += sc[t] * values[((long)t * 32 + b) * 256 + d];
    ctx[b * 256 + d] = f2bf(a);
  }
}

// ---------------------------------------------------------------------------
// Small utility kernels
// ---------------------------------------------------------------------------
__global__ void pack_w(bf16* __restrict__ dst, const float* __restrict__ src,
                       int R, int C, int Rp, int Cp)
{
  long i = (long)blockIdx.x * 256 + threadIdx.x;
  long total = (long)Rp * Cp;
  if (i >= total) return;
  int r = (int)(i / Cp), c = (int)(i % Cp);
  float v = (r < R && c < C) ? src[(long)r * C + c] : 0.f;
  dst[i] = f2bf(v);
}

__global__ void pack_x(bf16* __restrict__ dst, const float* __restrict__ x)
{
  // src (32,1024,40) -> dst ((t*32+b), 64) zero-padded
  long i = (long)blockIdx.x * 256 + threadIdx.x;
  if (i >= (long)32768 * 64) return;
  int c = (int)(i & 63);
  long row = i >> 6;
  int b = (int)(row & 31);
  int t = (int)(row >> 5);
  float v = (c < 40) ? x[((long)b * 1024 + t) * 40 + c] : 0.f;
  dst[i] = f2bf(v);
}

__global__ void gather_emb(bf16* __restrict__ dst, const float* __restrict__ tab,
                           const int* __restrict__ label)
{
  long i = (long)blockIdx.x * 256 + threadIdx.x;
  if (i >= (long)99 * 32 * 256) return;
  int c = (int)(i & 255);
  long row = i >> 8;
  int b = (int)(row & 31);
  int t = (int)(row >> 5);
  dst[i] = f2bf(tab[(long)label[b * 100 + t] * 256 + c]);
}

__global__ void combine_avg(bf16* __restrict__ dst, const bf16* __restrict__ a,
                            const bf16* __restrict__ b, long n)
{
  long i = (long)blockIdx.x * 256 + threadIdx.x;
  if (i < n) dst[i] = f2bf(0.5f * (bf2f(a[i]) + bf2f(b[i])));
}

__global__ void pyramid_pack(bf16* __restrict__ dst, const bf16* __restrict__ src,
                             int Th, int H)
{
  // src (2*Th,32,H) -> dst (Th,32,2H): concat(src[2t'], src[2t'+1]) on features
  long i = (long)blockIdx.x * 256 + threadIdx.x;
  long total = (long)Th * 32 * 2 * H;
  if (i >= total) return;
  int twoH = 2 * H;
  int c = (int)(i % twoH);
  long row = i / twoH;
  int b = (int)(row & 31);
  long tp = row >> 5;
  int half = (c >= H) ? 1 : 0;
  int cc = c - half * H;
  dst[i] = src[(((long)(2 * tp + half)) * 32 + b) * H + cc];
}

// ---------------------------------------------------------------------------
// Host orchestration
// ---------------------------------------------------------------------------
extern "C" void kernel_launch(void* const* d_in, const int* in_sizes, int n_in,
                              void* d_out, int out_size, void* d_ws, size_t ws_size,
                              hipStream_t stream)
{
  (void)in_sizes; (void)n_in; (void)out_size; (void)ws_size;
  const float* xin  = (const float*)d_in[0];
  const int* label  = (const int*)d_in[2];
  auto F = [&](int i) { return (const float*)d_in[i]; };

  // ---- workspace carve (256B aligned) ----
  char* base = (char*)d_ws;
  size_t off = 0;
  auto carve = [&](size_t bytes) -> char* {
    char* p = base + off;
    off += (bytes + 255) & ~(size_t)255;
    return p;
  };

  const int HH[4]   = {256, 512, 1024, 2048};
  const int IP[4]   = {64, 512, 1024, 2048};   // padded input dims
  const int Itru[4] = {40, 512, 1024, 2048};   // true input dims
  const int TT[4]   = {1024, 512, 256, 128};

  bf16 *encWih[4][2], *encWhh[4][2];
  for (int l = 0; l < 4; l++)
    for (int d = 0; d < 2; d++) {
      encWih[l][d] = (bf16*)carve((size_t)4 * HH[l] * IP[l] * 2);
      encWhh[l][d] = (bf16*)carve((size_t)4 * HH[l] * HH[l] * 2);
    }
  bf16* Wk_bf  = (bf16*)carve((size_t)256 * 2048 * 2);
  bf16* Wv_bf  = (bf16*)carve((size_t)256 * 2048 * 2);
  bf16* emb_bf = (bf16*)carve((size_t)1008 * 256 * 2);
  const int c4H[3] = {1024, 2048, 2048};
  const int cHd[3] = {256, 512, 512};
  bf16 *cWih[3], *cWhh[3];
  for (int i = 0; i < 3; i++) {
    cWih[i] = (bf16*)carve((size_t)c4H[i] * 512 * 2);
    cWhh[i] = (bf16*)carve((size_t)c4H[i] * cHd[i] * 2);
  }
  bf16* Wlin_bf = (bf16*)carve((size_t)256 * 256 * 2);
  bf16* W1_bf   = (bf16*)carve((size_t)256 * 512 * 2);

  bf16* xpad  = (bf16*)carve((size_t)32768 * 64 * 2);
  bf16* GxF   = (bf16*)carve((size_t)33554432 * 2);  // T*B x 4H (same all layers)
  bf16* GxB   = (bf16*)carve((size_t)33554432 * 2);
  bf16* hseqF = (bf16*)carve((size_t)8388608 * 2);   // T*B*H (same all layers)
  bf16* hseqB = (bf16*)carve((size_t)8388608 * 2);
  bf16* hcomb = (bf16*)carve((size_t)8388608 * 2);
  bf16* pyr   = (bf16*)carve((size_t)8388608 * 2);
  float* cF   = (float*)carve((size_t)32 * 2048 * 4);
  float* cB   = (float*)carve((size_t)32 * 2048 * 4);
  bf16* hzero = (bf16*)carve((size_t)32 * 2048 * 2);
  float* keys = (float*)carve((size_t)4096 * 256 * 4);
  float* vals = (float*)carve((size_t)4096 * 256 * 4);
  bf16* embSeq = (bf16*)carve((size_t)3168 * 256 * 2);
  bf16* embGx  = (bf16*)carve((size_t)3168 * 1024 * 2);
  float* qbuf  = (float*)carve((size_t)32 * 256 * 4);
  bf16* ctx_bf = (bf16*)carve((size_t)32 * 256 * 2);
  bf16* p_bf   = (bf16*)carve((size_t)32 * 256 * 2);
  bf16 *h0s[2], *h1s[2], *h2s[2];
  for (int i = 0; i < 2; i++) {
    h0s[i] = (bf16*)carve((size_t)32 * 256 * 2);
    h1s[i] = (bf16*)carve((size_t)32 * 512 * 2);
    h2s[i] = (bf16*)carve((size_t)32 * 512 * 2);
  }
  float* c0 = (float*)carve((size_t)32 * 256 * 4);
  float* c1 = (float*)carve((size_t)32 * 512 * 4);
  float* c2 = (float*)carve((size_t)32 * 512 * 4);

  auto packW = [&](bf16* dst, const float* src, int R, int C, int Rp, int Cp) {
    long total = (long)Rp * Cp;
    pack_w<<<dim3((unsigned)((total + 255) / 256)), 256, 0, stream>>>(dst, src, R, C, Rp, Cp);
  };

  // ---- pack weights to bf16 (padded) ----
  for (int l = 0; l < 4; l++)
    for (int d = 0; d < 2; d++) {
      int bi = 4 + l * 8 + d * 4;  // Wih, Whh, bih, bhh
      packW(encWih[l][d], F(bi + 0), 4 * HH[l], Itru[l], 4 * HH[l], IP[l]);
      packW(encWhh[l][d], F(bi + 1), 4 * HH[l], HH[l], 4 * HH[l], HH[l]);
    }
  packW(Wk_bf, F(36), 256, 2048, 256, 2048);
  packW(Wv_bf, F(38), 256, 2048, 256, 2048);
  packW(emb_bf, F(40), 1000, 256, 1008, 256);   // vocab padded 1000 -> 1008
  for (int i = 0; i < 3; i++) {
    packW(cWih[i], F(41 + i * 4 + 0), c4H[i], 512, c4H[i], 512);
    packW(cWhh[i], F(41 + i * 4 + 1), c4H[i], cHd[i], c4H[i], cHd[i]);
  }
  packW(Wlin_bf, F(53), 256, 256, 256, 256);
  packW(W1_bf, F(55), 256, 512, 256, 512);

  pack_x<<<dim3((32768 * 64) / 256), 256, 0, stream>>>(xpad, xin);
  hipMemsetAsync(hzero, 0, (size_t)32 * 2048 * 2, stream);

  // ---- encoder: 4 pyramidal BiLSTM layers ----
  const bf16* Xin = xpad;
  for (int l = 0; l < 4; l++) {
    int H = HH[l], T = TT[l], Ip = IP[l], fourH = 4 * H;
    for (int d = 0; d < 2; d++) {
      int bi = 4 + l * 8 + d * 4;
      bf16* Gx = d ? GxB : GxF;
      // Gx = Xin @ Wih^T + bih + bhh  (big WMMA GEMM over all timesteps)
      wmma_gemm<true><<<dim3(fourH / 64, (T * 32) / 32), 128, 0, stream>>>(
          Xin, Ip, encWih[l][d], Ip, F(bi + 2), F(bi + 3), Gx, fourH, Ip, fourH, fourH);
    }
    hipMemsetAsync(cF, 0, (size_t)32 * H * 4, stream);
    hipMemsetAsync(cB, 0, (size_t)32 * H * 4, stream);
    for (int s = 0; s < T; s++) {
      EncStepArgs a{GxF, GxB, encWhh[l][0], encWhh[l][1], hzero,
                    hseqF, hseqB, cF, cB, H, T, s};
      enc_step<<<dim3(H / 16, 2), 128, 0, stream>>>(a);
    }
    long cnt = (long)T * 32 * H;
    combine_avg<<<dim3((unsigned)((cnt + 255) / 256)), 256, 0, stream>>>(hcomb, hseqF, hseqB, cnt);
    if (l < 3) {
      pyramid_pack<<<dim3((unsigned)((cnt + 255) / 256)), 256, 0, stream>>>(pyr, hcomb, T / 2, H);
      Xin = pyr;
    }
  }

  // ---- keys / values: (128*32 x 2048) @ (256 x 2048)^T ----
  wmma_gemm<false><<<dim3(4, 128), 128, 0, stream>>>(
      hcomb, 2048, Wk_bf, 2048, F(37), nullptr, keys, 256, 2048, 256, 256);
  wmma_gemm<false><<<dim3(4, 128), 128, 0, stream>>>(
      hcomb, 2048, Wv_bf, 2048, F(39), nullptr, vals, 256, 2048, 256, 256);

  // ---- decoder precompute: embeddings + embedding half of cell0 gates ----
  gather_emb<<<dim3((99 * 32 * 256) / 256), 256, 0, stream>>>(embSeq, F(40), label);
  wmma_gemm<true><<<dim3(1024 / 64, 3168 / 32), 128, 0, stream>>>(
      embSeq, 256, cWih[0], 512, F(43), F(44), embGx, 1024, 256, 1024, 1024);
  hipMemsetAsync(c0, 0, (size_t)32 * 256 * 4, stream);
  hipMemsetAsync(c1, 0, (size_t)32 * 512 * 4, stream);
  hipMemsetAsync(c2, 0, (size_t)32 * 512 * 4, stream);
  for (int i = 0; i < 2; i++) {
    hipMemsetAsync(h0s[i], 0, (size_t)32 * 256 * 2, stream);
    hipMemsetAsync(h1s[i], 0, (size_t)32 * 512 * 2, stream);
    hipMemsetAsync(h2s[i], 0, (size_t)32 * 512 * 2, stream);
  }
  hipMemsetAsync(ctx_bf, 0, (size_t)32 * 256 * 2, stream);

  // ---- decoder: 99 steps ----
  float* outF = (float*)d_out;
  for (int t = 0; t < 99; t++) {
    int cur = t & 1, nxt = cur ^ 1;
    {  // cell0: gates = embGx[t] + ctx @ WihR^T + h0 @ Whh^T
      DecStepArgs a{};
      a.A0 = ctx_bf;   a.W0 = cWih[0] + 256; a.lda0 = 256; a.ldw0 = 512; a.K0 = 256;
      a.A1 = h0s[cur]; a.W1 = cWhh[0];       a.lda1 = 256; a.ldw1 = 256; a.K1 = 256;
      a.A2 = nullptr;  a.W2 = nullptr;       a.lda2 = 0;   a.ldw2 = 0;   a.K2 = 0;
      a.Gpre = embGx + (long)t * 32 * 1024;  a.ldg = 1024;
      a.b0 = nullptr; a.b1 = nullptr;
      a.c = c0; a.h = h0s[nxt]; a.H = 256;
      dec_step<<<dim3(16), 128, 0, stream>>>(a);
    }
    // q = h0 @ Wlin^T + blin
    wmma_gemm<false><<<dim3(4, 1), 128, 0, stream>>>(
        h0s[nxt], 256, Wlin_bf, 256, F(54), nullptr, qbuf, 256, 256, 256, 256);
    attention<<<dim3(32), 128, 0, stream>>>(qbuf, keys, vals, ctx_bf);
    {  // cell1: gates = h0 @ WihL^T + ctx @ WihR^T + h1 @ Whh^T + b
      DecStepArgs a{};
      a.A0 = h0s[nxt]; a.W0 = cWih[1];       a.lda0 = 256; a.ldw0 = 512; a.K0 = 256;
      a.A1 = ctx_bf;   a.W1 = cWih[1] + 256; a.lda1 = 256; a.ldw1 = 512; a.K1 = 256;
      a.A2 = h1s[cur]; a.W2 = cWhh[1];       a.lda2 = 512; a.ldw2 = 512; a.K2 = 512;
      a.Gpre = nullptr; a.ldg = 0;
      a.b0 = F(47); a.b1 = F(48);
      a.c = c1; a.h = h1s[nxt]; a.H = 512;
      dec_step<<<dim3(32), 128, 0, stream>>>(a);
    }
    {  // cell2: gates = h1 @ Wih^T + h2 @ Whh^T + b
      DecStepArgs a{};
      a.A0 = h1s[nxt]; a.W0 = cWih[2]; a.lda0 = 512; a.ldw0 = 512; a.K0 = 512;
      a.A1 = h2s[cur]; a.W1 = cWhh[2]; a.lda1 = 512; a.ldw1 = 512; a.K1 = 512;
      a.A2 = nullptr;  a.W2 = nullptr; a.lda2 = 0;   a.ldw2 = 0;   a.K2 = 0;
      a.Gpre = nullptr; a.ldg = 0;
      a.b0 = F(51); a.b1 = F(52);
      a.c = c2; a.h = h2s[nxt]; a.H = 512;
      dec_step<<<dim3(32), 128, 0, stream>>>(a);
    }
    // p = h2 @ W1^T + b1 (bf16 for tied projection)
    wmma_gemm<true><<<dim3(4, 1), 128, 0, stream>>>(
        h2s[nxt], 512, W1_bf, 512, F(56), nullptr, p_bf, 256, 512, 256, 256);
    // logits = p @ embed^T + b2 -> d_out[:, t, :]  (ldc = steps*V)
    wmma_gemm<false><<<dim3(16, 1), 128, 0, stream>>>(
        p_bf, 256, emb_bf, 256, F(57), nullptr, outF + (long)t * 1000, 99000L,
        256, 1000, 1008);
  }
}
